// Attention_65223373357517
// MI455X (gfx1250) — matrix operand order repaired and checked by reading, not compiled, not run on hardware
//
#include <hip/hip_runtime.h>

#define BB  16
#define LL  1024
#define DD  1024
#define DIN 4096
#define NEG_INF_F (-1e29f)
#define GELU_C 0.7978845608028654f

typedef __attribute__((ext_vector_type(16))) __bf16 v16bf;
typedef __attribute__((ext_vector_type(8)))  float  v8f;
typedef __attribute__((ext_vector_type(4)))  unsigned int u32x4;
typedef __attribute__((ext_vector_type(4)))  int          i32x4;

union FragBF { u32x4 u[2]; v16bf v; };

__device__ __forceinline__ unsigned short f32_to_bf16(float f) {
    unsigned int u = __float_as_uint(f);
    unsigned int r = u + 0x7FFFu + ((u >> 16) & 1u);   // round-to-nearest-even
    return (unsigned short)(r >> 16);
}
__device__ __forceinline__ float bf16_to_f32(unsigned short h) {
    return __uint_as_float(((unsigned int)h) << 16);
}

// ---------------------------------------------------------------------------
// CDNA5 async global->LDS copies (ASYNCcnt-tracked). Signature established by
// compile probe: (v4i AS1*, v4i AS3*, imm offset, imm cpol).
// ---------------------------------------------------------------------------
#if defined(__AMDGCN__) && \
    __has_builtin(__builtin_amdgcn_global_load_async_to_lds_b128) && \
    __has_builtin(__builtin_amdgcn_s_wait_asynccnt)
#define HAVE_ASYNC_LDS 1
#else
#define HAVE_ASYNC_LDS 0
#endif

#if HAVE_ASYNC_LDS
__device__ __forceinline__ void async_cp16(const unsigned short* g,
                                           unsigned short* l) {
    void* gv = (void*)g;   // drop const
    __builtin_amdgcn_global_load_async_to_lds_b128(
        (__attribute__((address_space(1))) i32x4*)gv,
        (__attribute__((address_space(3))) i32x4*)l, 0, 0);
}
#endif

// ---------------------------------------------------------------------------
// f32 -> bf16 convert
// ---------------------------------------------------------------------------
__global__ void cvt_bf16_kernel(const float* __restrict__ in,
                                unsigned short* __restrict__ out, size_t n) {
    size_t i = (size_t)blockIdx.x * blockDim.x + threadIdx.x;
    if (i < n) out[i] = f32_to_bf16(in[i]);
}

// ---------------------------------------------------------------------------
// bf16 transpose per batch: in [R,C] -> out [C,R]
// ---------------------------------------------------------------------------
__global__ __launch_bounds__(256) void transpose_bf16_kernel(
    const unsigned short* __restrict__ in, unsigned short* __restrict__ out,
    int R, int C) {
    __shared__ unsigned short tile[32][33];
    size_t boff = (size_t)blockIdx.z * R * C;
    in  += boff; out += boff;
    int r0 = blockIdx.y * 32, c0 = blockIdx.x * 32;
    int tx = threadIdx.x & 31, ty = threadIdx.x >> 5;   // 8 rows of 32
    #pragma unroll
    for (int i = ty; i < 32; i += 8)
        tile[i][tx] = in[(size_t)(r0 + i) * C + (c0 + tx)];
    __syncthreads();
    #pragma unroll
    for (int i = ty; i < 32; i += 8)
        out[(size_t)(c0 + i) * R + (r0 + tx)] = tile[tx][i];
}

// ---------------------------------------------------------------------------
// NT GEMM: C[M,N](f32) = A[M,K](bf16,row-major) * B[N,K](bf16,row-major)^T
// 128 threads = 4 waves (2x2), 64x64 per wave, 128x128 per block.
// Async global->LDS double-buffered staging; 16 WMMA per K-step of 32.
// Dims must be multiples of 128 (true for every call here).
// ---------------------------------------------------------------------------
__global__ __launch_bounds__(128) void gemm_nt_bf16_kernel(
    const unsigned short* __restrict__ A, const unsigned short* __restrict__ Bm,
    float* __restrict__ C, int M, int N, int K,
    size_t strideA, size_t strideB, size_t strideC) {
    int bz = blockIdx.z;
    A  += strideA * bz;
    Bm += strideB * bz;
    C  += strideC * bz;

    int lane = threadIdx.x & 31;
    int wave = threadIdx.x >> 5;
    int wm = (wave >> 1) * 64;               // wave tile inside 128x128 block
    int wn = (wave & 1) * 64;
    int bm = blockIdx.y * 128 + wm;
    int bn = blockIdx.x * 128 + wn;
    int half = lane >> 4;                    // 0: lanes 0-15, 1: lanes 16-31
    int lidx = lane & 15;

    v8f acc[4][4];
    #pragma unroll
    for (int mt = 0; mt < 4; ++mt)
        #pragma unroll
        for (int nt = 0; nt < 4; ++nt)
            acc[mt][nt] = {};

#if HAVE_ASYNC_LDS
    __shared__ unsigned short Asm[2][128 * 32];   // 8 KB per buffer
    __shared__ unsigned short Bsm[2][128 * 32];

    int t = threadIdx.x;
    int crow = t >> 2;            // copy row 0..31 (+ i*32)
    int ccol = (t & 3) * 8;       // copy col 0,8,16,24 (16B chunks)
    const unsigned short* Aband = A + (size_t)(blockIdx.y * 128) * K;
    const unsigned short* Bband = Bm + (size_t)(blockIdx.x * 128) * K;

    auto stage = [&](const unsigned short* G, unsigned short* L, int k0) {
        #pragma unroll
        for (int i = 0; i < 4; ++i)
            async_cp16(G + (size_t)(crow + i * 32) * K + k0 + ccol,
                       L + (crow + i * 32) * 32 + ccol);
    };
    stage(Aband, Asm[0], 0);
    stage(Bband, Bsm[0], 0);

    for (int k0 = 0; k0 < K; k0 += 32) {
        int cur = (k0 >> 5) & 1;
        __builtin_amdgcn_s_wait_asynccnt(0);   // buf[cur] landed (this thread)
        __syncthreads();                       // ... for every thread
        if (k0 + 32 < K) {                     // start filling buf[cur^1]
            stage(Aband, Asm[cur ^ 1], k0 + 32);
            stage(Bband, Bsm[cur ^ 1], k0 + 32);
        }

        FragBF af[4], bf_[4];
        // A fragment (16x32): lanes 0-15 row m, K {0..7,16..23};
        //                     lanes 16-31 row m, K {8..15,24..31}
        #pragma unroll
        for (int mt = 0; mt < 4; ++mt) {
            const unsigned short* p =
                &Asm[cur][(wm + mt * 16 + lidx) * 32 + half * 8];
            af[mt].u[0] = *(const u32x4*)p;
            af[mt].u[1] = *(const u32x4*)(p + 16);
        }
        // B fragment (32x16): lanes 0-15 col n, K {0..7,16..23}; lanes 16-31 K {8..15,24..31}
        #pragma unroll
        for (int nt = 0; nt < 4; ++nt) {
            const unsigned short* p =
                &Bsm[cur][(wn + nt * 16 + lidx) * 32 + half * 8];
            bf_[nt].u[0] = *(const u32x4*)p;
            bf_[nt].u[1] = *(const u32x4*)(p + 16);
        }
        #pragma unroll
        for (int mt = 0; mt < 4; ++mt)
            #pragma unroll
            for (int nt = 0; nt < 4; ++nt)
                acc[mt][nt] = __builtin_amdgcn_wmma_f32_16x16x32_bf16(
                    false, af[mt].v, false, bf_[nt].v,
                    (short)0, acc[mt][nt], false, false);
    }
#else
    for (int k0 = 0; k0 < K; k0 += 32) {
        FragBF af[4], bf_[4];
        #pragma unroll
        for (int mt = 0; mt < 4; ++mt) {
            const unsigned short* p =
                A + (size_t)(bm + mt * 16 + lidx) * K + k0 + half * 8;
            af[mt].u[0] = *(const u32x4*)p;
            af[mt].u[1] = *(const u32x4*)(p + 16);
            __builtin_prefetch(p + 64, 0, 3);
        }
        #pragma unroll
        for (int nt = 0; nt < 4; ++nt) {
            const unsigned short* p =
                Bm + (size_t)(bn + nt * 16 + lidx) * K + k0 + half * 8;
            bf_[nt].u[0] = *(const u32x4*)p;
            bf_[nt].u[1] = *(const u32x4*)(p + 16);
            __builtin_prefetch(p + 64, 0, 3);
        }
        #pragma unroll
        for (int mt = 0; mt < 4; ++mt)
            #pragma unroll
            for (int nt = 0; nt < 4; ++nt)
                acc[mt][nt] = __builtin_amdgcn_wmma_f32_16x16x32_bf16(
                    false, af[mt].v, false, bf_[nt].v,
                    (short)0, acc[mt][nt], false, false);
    }
#endif

    // C layout: VGPR r, lanes 0-15 -> M=r, lanes 16-31 -> M=8+r; N = lane&15
    #pragma unroll
    for (int mt = 0; mt < 4; ++mt)
        #pragma unroll
        for (int nt = 0; nt < 4; ++nt) {
            int m0 = bm + mt * 16 + half * 8;
            int c  = bn + nt * 16 + lidx;
            #pragma unroll
            for (int r = 0; r < 8; ++r)
                C[(size_t)(m0 + r) * N + c] = acc[mt][nt][r];
        }
}

// ---------------------------------------------------------------------------
// Masked row softmax over rows of length LL; writes bf16 probabilities.
// grid = (LL, BB), block = 256
// ---------------------------------------------------------------------------
__global__ __launch_bounds__(256) void softmax_row_kernel(
    const float* __restrict__ S, unsigned short* __restrict__ P,
    const int* __restrict__ rowmask, const int* __restrict__ colmask) {
    int b = blockIdx.y, row = blockIdx.x, tid = threadIdx.x;
    const float* srow = S + ((size_t)b * LL + row) * LL;
    unsigned short* prow = P + ((size_t)b * LL + row) * LL;
    float rm = (float)rowmask[b * LL + row];

    __shared__ float red[256];
    float v[4];
    float mx = -3.4e38f;
    #pragma unroll
    for (int i = 0; i < 4; ++i) {
        int c = tid + i * 256;
        float m = rm * (float)colmask[b * LL + c];
        v[i] = srow[c] + (1.0f - m) * NEG_INF_F;
        mx = fmaxf(mx, v[i]);
    }
    red[tid] = mx; __syncthreads();
    for (int s = 128; s > 0; s >>= 1) {
        if (tid < s) red[tid] = fmaxf(red[tid], red[tid + s]);
        __syncthreads();
    }
    mx = red[0]; __syncthreads();

    float sum = 0.0f;
    #pragma unroll
    for (int i = 0; i < 4; ++i) { v[i] = __expf(v[i] - mx); sum += v[i]; }
    red[tid] = sum; __syncthreads();
    for (int s = 128; s > 0; s >>= 1) {
        if (tid < s) red[tid] += red[tid + s];
        __syncthreads();
    }
    float inv = 1.0f / red[0];
    #pragma unroll
    for (int i = 0; i < 4; ++i)
        prow[tid + i * 256] = f32_to_bf16(v[i] * inv);
}

// ---------------------------------------------------------------------------
// Build heuristic input [x, y, x*y, x-y] as bf16 rows of length DIN
// ---------------------------------------------------------------------------
__global__ void build_inp_kernel(const float* __restrict__ x,
                                 const float* __restrict__ y,
                                 unsigned short* __restrict__ inp, size_t n) {
    size_t idx = (size_t)blockIdx.x * blockDim.x + threadIdx.x;
    if (idx >= n) return;
    size_t i = idx / DD;
    int    j = (int)(idx % DD);
    float xv = x[idx], yv = y[idx];
    unsigned short* row = inp + i * (size_t)DIN;
    row[j]            = f32_to_bf16(xv);
    row[DD + j]       = f32_to_bf16(yv);
    row[2 * DD + j]   = f32_to_bf16(xv * yv);
    row[3 * DD + j]   = f32_to_bf16(xv - yv);
}

// ---------------------------------------------------------------------------
// Final gating epilogue: out = sig(g) * gelu(r) + (1-sig(g)) * x
// ---------------------------------------------------------------------------
__global__ void gate_kernel(const float* __restrict__ rpre,
                            const float* __restrict__ gpre,
                            const float* __restrict__ br,
                            const float* __restrict__ bg,
                            const float* __restrict__ x,
                            float* __restrict__ out, size_t n) {
    size_t idx = (size_t)blockIdx.x * blockDim.x + threadIdx.x;
    if (idx >= n) return;
    int j = (int)(idx % DD);
    float r = rpre[idx] + br[j];
    float g = gpre[idx] + bg[j];
    float gel = 0.5f * r * (1.0f + tanhf(GELU_C * (r + 0.044715f * r * r * r)));
    float sig = 1.0f / (1.0f + __expf(-g));
    out[idx] = sig * gel + (1.0f - sig) * x[idx];
}

// ---------------------------------------------------------------------------
extern "C" void kernel_launch(void* const* d_in, const int* in_sizes, int n_in,
                              void* d_out, int out_size, void* d_ws, size_t ws_size,
                              hipStream_t stream) {
    const float* s   = (const float*)d_in[0];
    const float* q   = (const float*)d_in[1];
    const float* w_r = (const float*)d_in[2];
    const float* b_r = (const float*)d_in[3];
    const float* w_g = (const float*)d_in[4];
    const float* b_g = (const float*)d_in[5];
    const int* s_mask = (const int*)d_in[6];
    const int* q_mask = (const int*)d_in[7];
    float* out = (float*)d_out;

    const size_t EL  = (size_t)BB * LL * DD;        // 16M elements
    const size_t WEL = (size_t)DD * DIN;            // 4M (weight elements)
    const size_t SC  = (size_t)BB * LL * LL;        // 16M (score elements)
    const size_t INL = (size_t)BB * LL * DIN;       // 64M (heuristic input)

    char* ws = (char*)d_ws;
    size_t off = 0;
    auto alloc = [&](size_t bytes) -> char* {
        char* p = ws + off;
        off += (bytes + 255) & ~(size_t)255;
        return p;
    };
    unsigned short* s16    = (unsigned short*)alloc(EL * 2);
    unsigned short* q16    = (unsigned short*)alloc(EL * 2);
    unsigned short* sT16   = (unsigned short*)alloc(EL * 2);
    unsigned short* qT16   = (unsigned short*)alloc(EL * 2);
    unsigned short* wr16   = (unsigned short*)alloc(WEL * 2);
    unsigned short* wg16   = (unsigned short*)alloc(WEL * 2);
    float*          scA    = (float*)alloc(SC * 4);   // s·qᵀ
    float*          scB    = (float*)alloc(SC * 4);   // q·sᵀ
    unsigned short* p16    = (unsigned short*)alloc(SC * 2);
    unsigned short* p2_16  = (unsigned short*)alloc(SC * 2);
    float*          bmat   = (float*)alloc(EL * 4);
    float*          cmat   = (float*)alloc(EL * 4);
    unsigned short* inp_s  = (unsigned short*)alloc(INL * 2);
    unsigned short* inp_q  = (unsigned short*)alloc(INL * 2);
    float*          r_s    = (float*)alloc(EL * 4);
    float*          g_s    = (float*)alloc(EL * 4);
    float*          r_q    = (float*)alloc(EL * 4);
    float*          g_q    = (float*)alloc(EL * 4);
    (void)ws_size; (void)in_sizes; (void)n_in; (void)out_size;

    // 1) converts
    cvt_bf16_kernel<<<dim3((unsigned)(EL / 256)), 256, 0, stream>>>(s, s16, EL);
    cvt_bf16_kernel<<<dim3((unsigned)(EL / 256)), 256, 0, stream>>>(q, q16, EL);
    cvt_bf16_kernel<<<dim3((unsigned)(WEL / 256)), 256, 0, stream>>>(w_r, wr16, WEL);
    cvt_bf16_kernel<<<dim3((unsigned)(WEL / 256)), 256, 0, stream>>>(w_g, wg16, WEL);

    // 2) transposes (per batch [LL,DD] -> [DD,LL])
    transpose_bf16_kernel<<<dim3(DD / 32, LL / 32, BB), 256, 0, stream>>>(s16, sT16, LL, DD);
    transpose_bf16_kernel<<<dim3(DD / 32, LL / 32, BB), 256, 0, stream>>>(q16, qT16, LL, DD);

    // 3) scores: a = s·qᵀ ; aT = q·sᵀ   (batched NT GEMMs)
    gemm_nt_bf16_kernel<<<dim3(LL / 128, LL / 128, BB), 128, 0, stream>>>(
        s16, q16, scA, LL, LL, DD, (size_t)LL * DD, (size_t)LL * DD, (size_t)LL * LL);
    gemm_nt_bf16_kernel<<<dim3(LL / 128, LL / 128, BB), 128, 0, stream>>>(
        q16, s16, scB, LL, LL, DD, (size_t)LL * DD, (size_t)LL * DD, (size_t)LL * LL);

    // 4) masked softmax -> bf16 probabilities
    softmax_row_kernel<<<dim3(LL, BB), 256, 0, stream>>>(scA, p16, s_mask, q_mask);
    softmax_row_kernel<<<dim3(LL, BB), 256, 0, stream>>>(scB, p2_16, q_mask, s_mask);

    // 5) b = P·q (B operand = qT16 [DD,LL]) ; c = P2·s (B operand = sT16)
    gemm_nt_bf16_kernel<<<dim3(DD / 128, LL / 128, BB), 128, 0, stream>>>(
        p16, qT16, bmat, LL, DD, LL, (size_t)LL * LL, (size_t)LL * DD, (size_t)LL * DD);
    gemm_nt_bf16_kernel<<<dim3(DD / 128, LL / 128, BB), 128, 0, stream>>>(
        p2_16, sT16, cmat, LL, DD, LL, (size_t)LL * LL, (size_t)LL * DD, (size_t)LL * DD);

    // 6) heuristic inputs
    build_inp_kernel<<<dim3((unsigned)(EL / 256)), 256, 0, stream>>>(s, bmat, inp_s, EL);
    build_inp_kernel<<<dim3((unsigned)(EL / 256)), 256, 0, stream>>>(q, cmat, inp_q, EL);

    // 7) heuristic GEMMs: [B*L, DIN] x [DD, DIN]^T -> [B*L, DD]
    const int Mh = BB * LL;
    gemm_nt_bf16_kernel<<<dim3(DD / 128, Mh / 128, 1), 128, 0, stream>>>(
        inp_s, wr16, r_s, Mh, DD, DIN, 0, 0, 0);
    gemm_nt_bf16_kernel<<<dim3(DD / 128, Mh / 128, 1), 128, 0, stream>>>(
        inp_s, wg16, g_s, Mh, DD, DIN, 0, 0, 0);
    gemm_nt_bf16_kernel<<<dim3(DD / 128, Mh / 128, 1), 128, 0, stream>>>(
        inp_q, wr16, r_q, Mh, DD, DIN, 0, 0, 0);
    gemm_nt_bf16_kernel<<<dim3(DD / 128, Mh / 128, 1), 128, 0, stream>>>(
        inp_q, wg16, g_q, Mh, DD, DIN, 0, 0, 0);

    // 8) gating epilogues -> d_out = [s_tilde | q_tilde]
    gate_kernel<<<dim3((unsigned)(EL / 256)), 256, 0, stream>>>(
        r_s, g_s, b_r, b_g, s, out, EL);
    gate_kernel<<<dim3((unsigned)(EL / 256)), 256, 0, stream>>>(
        r_q, g_q, b_r, b_g, q, out + EL, EL);
}
